// PtrGenOutput_32023276159185
// MI455X (gfx1250) — matrix operand order repaired and checked by reading, not compiled, not run on hardware
//
#include <hip/hip_runtime.h>
#include <math.h>

// Problem constants (from reference)
#define Bsz   512
#define Hdim  512
#define Slen  200
#define OUTV  50000

typedef __attribute__((ext_vector_type(16))) _Float16 v16h;
typedef __attribute__((ext_vector_type(8)))  float    v8f;

// ---------------------------------------------------------------------------
// A-fragment (16x32 f16, ISA 7.12.2 "16-bit A-Matrix 16x32"):
// lanes 0-15 hold row M=lane with K=0..7 in elements 0..7 (V0-3) and
// K=16..23 in elements 8..15 (V4-7); lanes 16-31 hold K=8..15 / K=24..31.
// `p` must point at (row, k0 + kbA) where kbA = (lane>=16 ? 8 : 0).
// ---------------------------------------------------------------------------
__device__ __forceinline__ v16h load_fragA_f32(const float* __restrict__ p) {
  float4 c0 = *(const float4*)(p + 0);
  float4 c1 = *(const float4*)(p + 4);
  float4 c2 = *(const float4*)(p + 16);
  float4 c3 = *(const float4*)(p + 20);
  v16h v;
  v[0]  = (_Float16)c0.x; v[1]  = (_Float16)c0.y; v[2]  = (_Float16)c0.z; v[3]  = (_Float16)c0.w;
  v[4]  = (_Float16)c1.x; v[5]  = (_Float16)c1.y; v[6]  = (_Float16)c1.z; v[7]  = (_Float16)c1.w;
  v[8]  = (_Float16)c2.x; v[9]  = (_Float16)c2.y; v[10] = (_Float16)c2.z; v[11] = (_Float16)c2.w;
  v[12] = (_Float16)c3.x; v[13] = (_Float16)c3.y; v[14] = (_Float16)c3.z; v[15] = (_Float16)c3.w;
  return v;
}

// ---------------------------------------------------------------------------
// B-fragment (32x16 f16), same K order as A: lanes 0-15 hold N=lane with
// K=0..7 in elements 0..7 and K=16..23 in elements 8..15; lanes 16-31 hold
// N=lane-16 with K=8..15 / K=24..31.  `p` must point at (n_row, k0 + kbB)
// where kbB = (lane>=16 ? 8 : 0).
//
// ---------------------------------------------------------------------------
__device__ __forceinline__ v16h load_fragB_f32(const float* __restrict__ p) {
  float4 c0 = *(const float4*)(p + 0);
  float4 c1 = *(const float4*)(p + 4);
  float4 c2 = *(const float4*)(p + 16);
  float4 c3 = *(const float4*)(p + 20);
  v16h v;
  v[0]  = (_Float16)c0.x; v[1]  = (_Float16)c0.y; v[2]  = (_Float16)c0.z; v[3]  = (_Float16)c0.w;
  v[4]  = (_Float16)c1.x; v[5]  = (_Float16)c1.y; v[6]  = (_Float16)c1.z; v[7]  = (_Float16)c1.w;
  v[8]  = (_Float16)c2.x; v[9]  = (_Float16)c2.y; v[10] = (_Float16)c2.z; v[11] = (_Float16)c2.w;
  v[12] = (_Float16)c3.x; v[13] = (_Float16)c3.y; v[14] = (_Float16)c3.z; v[15] = (_Float16)c3.w;
  return v;
}

// ---------------------------------------------------------------------------
// GEMM: logits[b, n] = sum_k x[b,k] * W[n,k] + bias[n]
// Block: 256 threads = 8 waves arranged 4(M) x 2(N).
// Wave computes a 64x32 patch = 4x2 tiles of 16x16 via v_wmma_f32_16x16x32_f16.
// Grid: (ceil(OUTV/64), Bsz/256).
// ---------------------------------------------------------------------------
__global__ __launch_bounds__(256) void gemm_logits_kernel(
    const float* __restrict__ x, const float* __restrict__ W,
    const float* __restrict__ bias, float* __restrict__ logits) {
  const int lane = threadIdx.x & 31;
  const int wid  = threadIdx.x >> 5;
  const int wm   = wid & 3;
  const int wn   = wid >> 2;
  const int m0   = blockIdx.y * 256 + wm * 64;   // 4 m-tiles of 16 (always < 512)
  const int n0   = blockIdx.x * 64  + wn * 32;   // 2 n-tiles of 16
  const int lr   = lane & 15;
  const int kbA  = (lane >> 4) << 3;             // 0 or 8  (A: K chunk base)
  const int kbB  = (lane >> 4) << 3;             // 0 or 8  (B: K chunk base)

  v8f acc[4][2];
#pragma unroll
  for (int i = 0; i < 4; ++i)
#pragma unroll
    for (int j = 0; j < 2; ++j) acc[i][j] = (v8f){};

  for (int k0 = 0; k0 < Hdim; k0 += 32) {
    v16h af[4], bf[2];
#pragma unroll
    for (int i = 0; i < 4; ++i) {
      const int r = m0 + 16 * i + lr;
      af[i] = load_fragA_f32(x + (size_t)r * Hdim + k0 + kbA);
    }
#pragma unroll
    for (int j = 0; j < 2; ++j) {
      int n = n0 + 16 * j + lr;
      n = (n < OUTV) ? n : (OUTV - 1);           // clamp OOB loads, stores guarded
      bf[j] = load_fragB_f32(W + (size_t)n * Hdim + k0 + kbB);
    }
#pragma unroll
    for (int i = 0; i < 4; ++i)
#pragma unroll
      for (int j = 0; j < 2; ++j)
        acc[i][j] = __builtin_amdgcn_wmma_f32_16x16x32_f16(
            false, af[i], false, bf[j], (short)0, acc[i][j], false, false);
  }

  // Epilogue: C/D layout (ISA 7.12.2): VGPR r, lanes 0-15 -> M=r, lanes 16-31 -> M=8+r.
  const int rb = (lane >> 4) << 3;               // 0 or 8
#pragma unroll
  for (int j = 0; j < 2; ++j) {
    const int col = n0 + 16 * j + lr;
    if (col >= OUTV) continue;
    const float bv = bias[col];
#pragma unroll
    for (int i = 0; i < 4; ++i) {
      const int row = m0 + 16 * i + rb;
#pragma unroll
      for (int r = 0; r < 8; ++r)
        logits[(size_t)(row + r) * OUTV + col] = acc[i][j][r] + bv;
    }
  }
}

// ---------------------------------------------------------------------------
// Gate (B,2) log-softmax + attention (B,S) softmax. One block per row.
// ---------------------------------------------------------------------------
__global__ __launch_bounds__(256) void cog_attn_kernel(
    const float* __restrict__ x, const float* __restrict__ cogW,
    const float* __restrict__ cogb, const float* __restrict__ attn_s,
    float* __restrict__ pog, float* __restrict__ attn_p) {
  __shared__ float r0[256];
  __shared__ float r1[256];
  const int b = blockIdx.x;
  const int t = threadIdx.x;

  // cog dot products over H
  float a0 = 0.f, a1 = 0.f;
  for (int h = t; h < Hdim; h += 256) {
    const float xv = x[(size_t)b * Hdim + h];
    a0 += xv * cogW[h];
    a1 += xv * cogW[Hdim + h];
  }
  r0[t] = a0; r1[t] = a1;
  __syncthreads();
  for (int s = 128; s > 0; s >>= 1) {
    if (t < s) { r0[t] += r0[t + s]; r1[t] += r1[t + s]; }
    __syncthreads();
  }
  if (t == 0) {
    const float l0 = r0[0] + cogb[0];
    const float l1 = r1[0] + cogb[1];
    const float m  = fmaxf(l0, l1);
    const float lse = m + logf(expf(l0 - m) + expf(l1 - m));
    pog[2 * b]     = l0 - lse;
    pog[2 * b + 1] = l1 - lse;
  }
  __syncthreads();

  // attention softmax over S
  const float v = (t < Slen) ? attn_s[(size_t)b * Slen + t] : -INFINITY;
  r0[t] = v;
  __syncthreads();
  for (int s = 128; s > 0; s >>= 1) {
    if (t < s) r0[t] = fmaxf(r0[t], r0[t + s]);
    __syncthreads();
  }
  const float mx = r0[0];
  __syncthreads();
  const float e = (t < Slen) ? expf(v - mx) : 0.f;
  r0[t] = e;
  __syncthreads();
  for (int s = 128; s > 0; s >>= 1) {
    if (t < s) r0[t] += r0[t + s];
    __syncthreads();
  }
  if (t < Slen) attn_p[(size_t)b * Slen + t] = e / r0[0];
}

// ---------------------------------------------------------------------------
// Row-wise log-softmax over OUTV with out_map gather; writes gen_probs and
// the default out_probs = g0 + gen_probs (ptr path is -inf a.e.).
// Raw logits live in the gen_probs region of d_out (overwritten in place;
// out_map is a permutation -> each thread reads its own index before writing).
// ---------------------------------------------------------------------------
__global__ __launch_bounds__(256) void row_logsoftmax_kernel(
    const float* __restrict__ logits, const int* __restrict__ out_map,
    const float* __restrict__ pog, float* __restrict__ gen_p,
    float* __restrict__ out_p) {
  const int b = blockIdx.x;
  const int t = threadIdx.x;
  const float* row = logits + (size_t)b * OUTV;

  // online logsumexp
  float m = -INFINITY, s = 0.f;
  for (int j = t; j < OUTV; j += 256) {
    const float v = row[out_map[j]];
    if (v > m) { s = s * expf(m - v) + 1.f; m = v; }
    else       { s += expf(v - m); }
  }
  __shared__ float sm[256];
  __shared__ float ss[256];
  sm[t] = m; ss[t] = s;
  __syncthreads();
  for (int st = 128; st > 0; st >>= 1) {
    if (t < st) {
      const float mA = sm[t], mB = sm[t + st];
      const float M = fmaxf(mA, mB);
      const float sa = (mA == -INFINITY) ? 0.f : ss[t] * expf(mA - M);
      const float sb = (mB == -INFINITY) ? 0.f : ss[t + st] * expf(mB - M);
      sm[t] = M; ss[t] = sa + sb;
    }
    __syncthreads();
  }
  const float lse = sm[0] + logf(ss[0]);
  const float g0  = pog[2 * b];
  for (int j = t; j < OUTV; j += 256) {
    const float gp = row[out_map[j]] - lse;
    gen_p[(size_t)b * OUTV + j] = gp;
    out_p[(size_t)b * OUTV + j] = g0 + gp;
  }
}

// ---------------------------------------------------------------------------
// Sparse copy-path fixup: per row, scatter-add attn into per-v sums, then
// `.set` onto output ids (later v wins on collision, matching one legal
// resolution of the reference's duplicate-index scatter), then patch
// out_probs with stable logaddexp.  One thread per row.
// ---------------------------------------------------------------------------
__global__ void ptr_fixup_kernel(
    const int* __restrict__ inptensor, const int* __restrict__ inp_to_act,
    const float* __restrict__ attn_p, const float* __restrict__ pog,
    const float* __restrict__ gen_p, float* __restrict__ out_p) {
  const int b = blockIdx.x * blockDim.x + threadIdx.x;
  if (b >= Bsz) return;

  int   vids[Slen]; float vsum[Slen]; int nv = 0;
  for (int s = 0; s < Slen; ++s) {
    const int v   = inptensor[(size_t)b * Slen + s];
    const float p = attn_p[(size_t)b * Slen + s];
    int f = -1;
    for (int i = 0; i < nv; ++i) { if (vids[i] == v) { f = i; break; } }
    if (f >= 0) vsum[f] += p;
    else { vids[nv] = v; vsum[nv] = p; ++nv; }
  }

  int aids[Slen]; float aval[Slen]; int na = 0;
  for (int i = 0; i < nv; ++i) {
    const int a = inp_to_act[vids[i]];
    int f = -1;
    for (int j = 0; j < na; ++j) { if (aids[j] == a) { f = j; break; } }
    if (f >= 0) aval[f] = vsum[i];                  // set semantics
    else { aids[na] = a; aval[na] = vsum[i]; ++na; }
  }

  const float g0 = pog[2 * b];
  const float g1 = pog[2 * b + 1];
  for (int j = 0; j < na; ++j) {
    const size_t idx = (size_t)b * OUTV + aids[j];
    const float t0 = g0 + gen_p[idx];
    const float t1 = g1 + logf(aval[j]);
    const float mx = fmaxf(t0, t1);
    const float mn = fminf(t0, t1);
    out_p[idx] = mx + log1pf(expf(mn - mx));
  }
}

// ---------------------------------------------------------------------------
extern "C" void kernel_launch(void* const* d_in, const int* in_sizes, int n_in,
                              void* d_out, int out_size, void* d_ws, size_t ws_size,
                              hipStream_t stream) {
  (void)in_sizes; (void)n_in; (void)out_size; (void)d_ws; (void)ws_size;

  const float* x          = (const float*)d_in[0];
  const float* gen_W      = (const float*)d_in[1];
  const float* gen_b      = (const float*)d_in[2];
  const float* cog_W      = (const float*)d_in[3];
  const float* cog_b      = (const float*)d_in[4];
  const float* attn_s     = (const float*)d_in[5];
  const int*   inptensor  = (const int*)d_in[6];
  const int*   inp_to_act = (const int*)d_in[7];
  const int*   out_map    = (const int*)d_in[8];

  // d_out layout (flat, return order): out_probs | ptr_or_gen | gen_probs | attn_probs
  float* out_p  = (float*)d_out;
  float* pog    = out_p + (size_t)Bsz * OUTV;
  float* gen_p  = pog + (size_t)Bsz * 2;            // also raw-logit scratch
  float* attn_p = gen_p + (size_t)Bsz * OUTV;

  // 1) gate + attention softmax
  cog_attn_kernel<<<Bsz, 256, 0, stream>>>(x, cog_W, cog_b, attn_s, pog, attn_p);

  // 2) WMMA GEMM -> raw logits into gen_probs region
  dim3 ggrid((OUTV + 63) / 64, Bsz / 256);
  gemm_logits_kernel<<<ggrid, 256, 0, stream>>>(x, gen_W, gen_b, gen_p);

  // 3) row log-softmax (+ out_map gather) -> gen_probs (in place) and default out_probs
  row_logsoftmax_kernel<<<Bsz, 256, 0, stream>>>(gen_p, out_map, pog, gen_p, out_p);

  // 4) sparse copy-path fixup of out_probs
  ptr_fixup_kernel<<<(Bsz + 127) / 128, 128, 0, stream>>>(
      inptensor, inp_to_act, attn_p, pog, gen_p, out_p);
}
